// SymbolicPenaltyTracker_67594195304468
// MI455X (gfx1250) — compile-verified, matrix operand order checked by reading, not run on hardware
//
#include <hip/hip_runtime.h>
#include <hip/hip_bf16.h>
#include <stdint.h>

typedef __attribute__((ext_vector_type(16))) __bf16 v16bf;
typedef __attribute__((ext_vector_type(8)))  float  v8f;

#define T_DIM 4096
#define D_DIM 256
#define BM 128
#define BK 32
#define LDA 40   // ushort row stride for A tile (padded: conflict-free b128 frag reads)
#define LDK 40   // ushort row stride for transposed k tile

// f32 pair -> packed bf16, round-to-nearest (ties away): add 0x8000 then take the
// high halves of both words with a single v_perm_b32. 3 VALU per pair.
__device__ __forceinline__ unsigned int pk_bf16(float lo, float hi) {
  unsigned int a = __float_as_uint(lo) + 0x8000u;
  unsigned int b = __float_as_uint(hi) + 0x8000u;
  // perm(S0=b, S1=a): sel bytes 0x02,0x03 = a[2],a[3]; 0x06,0x07 = b[2],b[3]
  return __builtin_amdgcn_perm(b, a, 0x07060302u);
}

union FragU { struct { uint4 a, b; } u; v16bf v; };

__device__ __forceinline__ v16bf load_frag(const unsigned short* p) {
  // ISA 16-bit A/B layout: per lane, halves 0..7 = K[kb..kb+7], halves 8..15 = K[kb+16..kb+23]
  FragU f;
  f.u.a = *(const uint4*)(p);
  f.u.b = *(const uint4*)(p + 16);
  return f.v;
}

__global__ __launch_bounds__(256)
void spt_rowsum(const float* __restrict__ A, float* __restrict__ dinv) {
  const int lane = threadIdx.x & 31;
  const int row  = blockIdx.x * 8 + (threadIdx.x >> 5);   // flattened b*T + t
  const float4* ar = (const float4*)(A + (size_t)row * T_DIM);
  float s = 0.f;
#pragma unroll 8
  for (int j = 0; j < 32; ++j) {
    float4 v = ar[j * 32 + lane];
    s += (v.x + v.y) + (v.z + v.w);
  }
#pragma unroll
  for (int off = 16; off > 0; off >>= 1) s += __shfl_xor(s, off, 32);
  if (lane == 0) dinv[row] = rsqrtf(s + 1e-6f);
}

template <bool MASK>
__device__ __forceinline__ void gemm_step(
    int s0, const float* __restrict__ a_row, const float* __restrict__ dinv_b,
    float dinv_t, int tlim, int ar, int ah,
    int grp, int lane64, const float* __restrict__ k_b,
    unsigned short* lds_a, unsigned short* lds_k,
    const unsigned short* afp, int l15, int kb,
    v8f acc[16], float& rmax)
{
  __syncthreads();

  // --- build W tile: scale, (causal mask), row-max, pack bf16 ---
#pragma unroll
  for (int j = 0; j < 4; ++j) {
    const int c = ah + j * 4;
    float4 av = *(const float4*)(a_row + s0 + c);
    float4 dv = *(const float4*)(dinv_b + s0 + c);
    float w0 = av.x * dinv_t * dv.x;
    float w1 = av.y * dinv_t * dv.y;
    float w2 = av.z * dinv_t * dv.z;
    float w3 = av.w * dinv_t * dv.w;
    if (MASK) {
      if (s0 + c + 0 > tlim) w0 = 0.f;
      if (s0 + c + 1 > tlim) w1 = 0.f;
      if (s0 + c + 2 > tlim) w2 = 0.f;
      if (s0 + c + 3 > tlim) w3 = 0.f;
    }
    rmax = fmaxf(rmax, fmaxf(fmaxf(w0, w1), fmaxf(w2, w3)));
    *(unsigned int*)&lds_a[ar * LDA + c + 0] = pk_bf16(w0, w1);
    *(unsigned int*)&lds_a[ar * LDA + c + 2] = pk_bf16(w2, w3);
  }

  // --- build k tile transposed: [d][s] bf16, pack s-pairs into b32 stores ---
#pragma unroll
  for (int j = 0; j < 4; ++j) {
    const int sl = grp * 8 + j * 2;
    const float* kr0 = k_b + (size_t)(s0 + sl) * D_DIM + lane64 * 4;
    float4 ka = *(const float4*)(kr0);
    float4 kc = *(const float4*)(kr0 + D_DIM);
    const int d0 = lane64 * 4;
    *(unsigned int*)&lds_k[(d0 + 0) * LDK + sl] = pk_bf16(ka.x, kc.x);
    *(unsigned int*)&lds_k[(d0 + 1) * LDK + sl] = pk_bf16(ka.y, kc.y);
    *(unsigned int*)&lds_k[(d0 + 2) * LDK + sl] = pk_bf16(ka.z, kc.z);
    *(unsigned int*)&lds_k[(d0 + 3) * LDK + sl] = pk_bf16(ka.w, kc.w);
  }
  __syncthreads();

  // --- 16 WMMAs per wave per K-step, one shared A fragment ---
  v16bf afrag = load_frag(afp);
#pragma unroll
  for (int nt = 0; nt < 16; ++nt) {
    v16bf bfrag = load_frag(&lds_k[(nt * 16 + l15) * LDK + kb]);
    acc[nt] = __builtin_amdgcn_wmma_f32_16x16x32_bf16(
        false, afrag, false, bfrag, (short)0, acc[nt], false, false);
  }
}

__global__ __launch_bounds__(256)
void spt_gemm(const float* __restrict__ A, const float* __restrict__ Kq,
              const float* __restrict__ dinv, float* __restrict__ out) {
  __shared__ unsigned short lds_a[BM * LDA];      // W tile, bf16, [m][k]
  __shared__ unsigned short lds_k[D_DIM * LDK];   // k tile transposed, bf16, [d][k]
  __shared__ int lds_rmax[BM];

  const int tid = threadIdx.x;
  const int b   = blockIdx.y;
  const int t0  = blockIdx.x * BM;

  if (tid < BM) lds_rmax[tid] = 0;

  // A-tile producer mapping: 2 threads per row, 16 cols each
  const int ar = tid >> 1;
  const int ah = (tid & 1) * 16;
  const float* dinv_b = dinv + b * T_DIM;
  const float  dinv_t = dinv_b[t0 + ar];
  const float* a_row  = A + ((size_t)(b * T_DIM + t0 + ar)) * T_DIM;
  const int tlim = t0 + ar;
  float rmax = 0.f;

  // k-tile producer mapping: 4 groups x 64 lanes; each handles 8 s-rows, 4 d-cols
  const int lane64 = tid & 63;
  const int grp    = tid >> 6;
  const float* k_b = Kq + (size_t)b * T_DIM * D_DIM;

  // WMMA consumer mapping
  const int lane = tid & 31;
  const int wv   = tid >> 5;
  const int l15  = lane & 15;
  const int kb   = (lane < 16) ? 0 : 8;
  const unsigned short* afp = &lds_a[(wv * 16 + l15) * LDA + kb];

  v8f acc[16];
  v8f zero = {0.f, 0.f, 0.f, 0.f, 0.f, 0.f, 0.f, 0.f};
#pragma unroll
  for (int i = 0; i < 16; ++i) acc[i] = zero;

  const int full   = t0 / BK;          // off-diagonal K-steps: no causal mask needed
  const int nsteps = full + BM / BK;   // + 4 diagonal steps

  for (int step = 0; step < full; ++step)
    gemm_step<false>(step * BK, a_row, dinv_b, dinv_t, tlim, ar, ah,
                     grp, lane64, k_b, lds_a, lds_k, afp, l15, kb, acc, rmax);
  for (int step = full; step < nsteps; ++step)
    gemm_step<true>(step * BK, a_row, dinv_b, dinv_t, tlim, ar, ah,
                    grp, lane64, k_b, lds_a, lds_k, afp, l15, kb, acc, rmax);

  atomicMax(&lds_rmax[ar], __float_as_int(rmax));   // W >= 0: int order == float order
  __syncthreads();

  // C/D layout: VGPR r, lanes 0-15 -> (M=r, N=lane); lanes 16-31 -> (M=r+8, N=lane-16)
  const int rloc = wv * 16 + ((lane < 16) ? 0 : 8);
#pragma unroll
  for (int r = 0; r < 8; ++r) {
    float m  = __int_as_float(lds_rmax[rloc + r]);
    float sc = (m > 1e-9f) ? 0.70710678118654752f : 0.f;
    float* orow = out + ((size_t)(b * T_DIM + t0 + rloc + r)) * D_DIM + l15;
#pragma unroll
    for (int nt = 0; nt < 16; ++nt) orow[nt * 16] = acc[nt][r] * sc;
  }
}

extern "C" void kernel_launch(void* const* d_in, const int* in_sizes, int n_in,
                              void* d_out, int out_size, void* d_ws, size_t ws_size,
                              hipStream_t stream) {
  const float* A  = (const float*)d_in[0];
  const float* Kq = (const float*)d_in[1];
  float* out  = (float*)d_out;
  float* dinv = (float*)d_ws;                      // B*T floats of scratch
  const int B = in_sizes[1] / (T_DIM * D_DIM);

  spt_rowsum<<<dim3(B * T_DIM / 8), 256, 0, stream>>>(A, dinv);
  spt_gemm<<<dim3(T_DIM / BM, B), 256, 0, stream>>>(A, Kq, dinv, out);
}
